// MLGRUCell_21706764714146
// MI455X (gfx1250) — 8-bit matrix operand order repaired and checked by reading, compiled, not run on hardware
//
#include <hip/hip_runtime.h>

typedef __attribute__((ext_vector_type(8))) int          v8i;
typedef __attribute__((ext_vector_type(4))) float        v4f;
typedef __attribute__((ext_vector_type(4))) unsigned int v4u;

#define NROWS          262144
#define DIM            128
#define ROWS_PER_BLOCK 256
#define ITERS          2
#define NTHREADS       256
#define NWAVES         8
#define EPSV           1e-8f

static __device__ __forceinline__ v8i lds_load_frag(const unsigned int* p) {
  v4u a = *(const v4u*)(p);
  v4u b = *(const v4u*)(p + 4);
  v8i r;
  r[0] = (int)a.x; r[1] = (int)a.y; r[2] = (int)a.z; r[3] = (int)a.w;
  r[4] = (int)b.x; r[5] = (int)b.y; r[6] = (int)b.z; r[7] = (int)b.w;
  return r;
}

static __device__ __forceinline__ float sigmoid_f(float x) {
  return 1.0f / (1.0f + __expf(-x));
}

__global__ __launch_bounds__(NTHREADS)
void mlgru_wmma_kernel(const float* __restrict__ x_t,
                       const float* __restrict__ h_prev,
                       const float* __restrict__ W_f,
                       const float* __restrict__ W_c,
                       const float* __restrict__ W_g,
                       const float* __restrict__ b_f,
                       const float* __restrict__ b_c,
                       const float* __restrict__ b_g,
                       float* __restrict__ o_out,
                       float* __restrict__ h_out) {
  __shared__ __align__(16) unsigned int ldsB[48 * 256];
  __shared__ __align__(16) unsigned int ldsA[NWAVES * 512];
  __shared__ float ldsSv[NWAVES * 16];

  const int tid  = threadIdx.x;
  const int wave = tid >> 5;
  const int lane = tid & 31;

  const float* Wm0 = W_f;
  const float* Wm1 = W_c;
  const float* Wm2 = W_g;
  for (int idx = tid; idx < 48 * 256; idx += NTHREADS) {
    int F = idx >> 8;
    int w = idx & 255;
    int l = w >> 3;
    int v = w & 7;
    int m = F >> 4;
    int t = (F >> 1) & 7;
    int h = F & 1;
    int n  = (l & 15) + t * 16;
    int kb = h * 64 + (v >> 1) * 16 + (l >> 4) * 8 + (v & 1) * 4;
    const float* wp = (m == 0 ? Wm0 : (m == 1 ? Wm1 : Wm2)) + n * DIM + kb;
    unsigned int pack = 0;
#pragma unroll
    for (int i = 0; i < 4; ++i) {
      float wv = wp[i];
      int sg = (wv > 0.0f) ? 1 : ((wv < 0.0f) ? -1 : 0);
      pack |= ((unsigned int)((unsigned char)(signed char)sg)) << (8 * i);
    }
    ldsB[idx] = pack;
  }
  __syncthreads();

  unsigned int* myA  = ldsA + wave * 512;
  float*        mySv = ldsSv + wave * 16;

  const int K0    = lane * 4;
  const int fragW = K0 >> 6;
  const int kin   = K0 & 63;
  const int kk    = kin & 15;
  const int dstV  = (kin >> 4) * 2 + ((kk & 7) >> 2);
  const int dstHi = (kk >> 3) << 4;

  const int hb  = (lane >> 4) * 8;
  const int col = lane & 15;

  for (int iter = 0; iter < ITERS; ++iter) {
    const int row0 = blockIdx.x * ROWS_PER_BLOCK + iter * 128 + wave * 16;

    if (iter + 1 < ITERS) {
      __builtin_prefetch(x_t + (size_t)(row0 + 128) * DIM + lane * 64, 0, 0);
    }

    for (int r = 0; r < 16; ++r) {
      const float* xr = x_t + (size_t)(row0 + r) * DIM;
      v4f xv = *(const v4f*)(xr + lane * 4);
      float s1 = xv.x + xv.y + xv.z + xv.w;
      float s2 = xv.x * xv.x + xv.y * xv.y + xv.z * xv.z + xv.w * xv.w;
#pragma unroll
      for (int off = 16; off > 0; off >>= 1) {
        s1 += __shfl_xor(s1, off, 32);
        s2 += __shfl_xor(s2, off, 32);
      }
      float mean = s1 * (1.0f / 128.0f);
      float var  = s2 * (1.0f / 128.0f) - mean * mean;
      float rstd = rsqrtf(var + EPSV);
      float x0 = (xv.x - mean) * rstd;
      float x1 = (xv.y - mean) * rstd;
      float x2 = (xv.z - mean) * rstd;
      float x3 = (xv.w - mean) * rstd;
      float am = fmaxf(fmaxf(fabsf(x0), fabsf(x1)), fmaxf(fabsf(x2), fabsf(x3)));
#pragma unroll
      for (int off = 16; off > 0; off >>= 1) {
        am = fmaxf(am, __shfl_xor(am, off, 32));
      }
      float s  = 127.0f / am;
      int q0 = (int)fminf(fmaxf(rintf(s * x0), -128.0f), 127.0f);
      int q1 = (int)fminf(fmaxf(rintf(s * x1), -128.0f), 127.0f);
      int q2 = (int)fminf(fmaxf(rintf(s * x2), -128.0f), 127.0f);
      int q3 = (int)fminf(fmaxf(rintf(s * x3), -128.0f), 127.0f);
      unsigned int pack = (unsigned int)(q0 & 255) |
                          ((unsigned int)(q1 & 255) << 8) |
                          ((unsigned int)(q2 & 255) << 16) |
                          ((unsigned int)(q3 & 255) << 24);
      myA[fragW * 256 + (r + dstHi) * 8 + dstV] = pack;
      if (lane == 0) mySv[r] = am * (1.0f / 127.0f);
    }

    v8i A0 = lds_load_frag(myA + lane * 8);
    v8i A1 = lds_load_frag(myA + 256 + lane * 8);
    float sv[8];
#pragma unroll
    for (int j = 0; j < 8; ++j) sv[j] = mySv[hb + j];

    for (int t = 0; t < 8; ++t) {
      const unsigned int* bt = ldsB + t * 512 + lane * 8;
      v8i accF = {0, 0, 0, 0, 0, 0, 0, 0};
      v8i accC = {0, 0, 0, 0, 0, 0, 0, 0};
      v8i accG = {0, 0, 0, 0, 0, 0, 0, 0};
      {
        v8i B0 = lds_load_frag(bt);
        v8i B1 = lds_load_frag(bt + 256);
        accF = __builtin_amdgcn_wmma_i32_16x16x64_iu8(true, A0, true, B0, accF, false, false);
        accF = __builtin_amdgcn_wmma_i32_16x16x64_iu8(true, A1, true, B1, accF, false, false);
      }
      {
        v8i B0 = lds_load_frag(bt + 1 * 4096);
        v8i B1 = lds_load_frag(bt + 1 * 4096 + 256);
        accC = __builtin_amdgcn_wmma_i32_16x16x64_iu8(true, A0, true, B0, accC, false, false);
        accC = __builtin_amdgcn_wmma_i32_16x16x64_iu8(true, A1, true, B1, accC, false, false);
      }
      {
        v8i B0 = lds_load_frag(bt + 2 * 4096);
        v8i B1 = lds_load_frag(bt + 2 * 4096 + 256);
        accG = __builtin_amdgcn_wmma_i32_16x16x64_iu8(true, A0, true, B0, accG, false, false);
        accG = __builtin_amdgcn_wmma_i32_16x16x64_iu8(true, A1, true, B1, accG, false, false);
      }

      float bfv = b_f[t * 16 + col];
      float bcv = b_c[t * 16 + col];
      float bgv = b_g[t * 16 + col];

      const size_t gbase = (size_t)row0 * DIM + (size_t)(t * 16 + col);
#pragma unroll
      for (int j = 0; j < 8; ++j) {
        float yf = (float)accF[j] * sv[j] + bfv;
        float yc = (float)accC[j] * sv[j] + bcv;
        float yg = (float)accG[j] * sv[j] + bgv;
        float f = sigmoid_f(yf);
        float c = yc * sigmoid_f(yc);
        float g = sigmoid_f(yg);
        size_t off = gbase + (size_t)(hb + j) * DIM;
        float hp = h_prev[off];
        float ht = f * hp + (1.0f - f) * c;
        o_out[off] = g * ht;
        h_out[off] = ht;
      }
    }
  }
}

extern "C" void kernel_launch(void* const* d_in, const int* in_sizes, int n_in,
                              void* d_out, int out_size, void* d_ws, size_t ws_size,
                              hipStream_t stream) {
  const float* x_t    = (const float*)d_in[0];
  const float* h_prev = (const float*)d_in[1];
  const float* W_f    = (const float*)d_in[2];
  const float* W_c    = (const float*)d_in[3];
  const float* W_g    = (const float*)d_in[4];
  const float* b_f    = (const float*)d_in[5];
  const float* b_c    = (const float*)d_in[6];
  const float* b_g    = (const float*)d_in[7];

  float* o_out = (float*)d_out;
  float* h_out = o_out + (size_t)NROWS * DIM;

  dim3 grid(NROWS / ROWS_PER_BLOCK);
  dim3 block(NTHREADS);
  mlgru_wmma_kernel<<<grid, block, 0, stream>>>(
      x_t, h_prev, W_f, W_c, W_g, b_f, b_c, b_g, o_out, h_out);
}
